// WanSelfAttention_75857712382562
// MI455X (gfx1250) — matrix operand order repaired and checked by reading, not compiled, not run on hardware
//
#include <hip/hip_runtime.h>

// ---------------------------------------------------------------------------
// Problem constants (match reference module)
// ---------------------------------------------------------------------------
#define DIM    1536
#define N_H    12
#define D_H    128
#define FS     300
#define SEQ    2400         // S = 8*FS
#define KEYS   4200         // end_idx == CACHE
#define KEYP   4224         // KEYS padded to multiple of 32 for WMMA tiling
#define C_HALF 64
#define C0     22
#define C1     21
#define ATT_SCALE 0.088388347648318447f   // 1/sqrt(128)

typedef __attribute__((ext_vector_type(16))) _Float16 v16h;
typedef __attribute__((ext_vector_type(8)))  _Float16 v8h;
typedef __attribute__((ext_vector_type(8)))  float    v8f;

__device__ __forceinline__ v16h cat8(v8h lo, v8h hi) {
    return __builtin_shufflevector(lo, hi, 0,1,2,3,4,5,6,7,8,9,10,11,12,13,14,15);
}

// ---------------------------------------------------------------------------
// Async global->LDS staging (CDNA5 GLOBAL_LOAD_ASYNC_TO_LDS_B128, ASYNCcnt),
// guarded so a toolchain without the builtin still compiles (sync fallback).
// Builtin signature (from clang diagnostic): (int4 as1*, int4 as3*, imm, imm).
// ---------------------------------------------------------------------------
#if defined(__has_builtin)
#  if __has_builtin(__builtin_amdgcn_global_load_async_to_lds_b128)
#    define HAVE_ASYNC_LDS 1
#  endif
#endif

#ifdef HAVE_ASYNC_LDS
typedef int v4i_vs __attribute__((vector_size(4 * sizeof(int))));
typedef __attribute__((address_space(1))) v4i_vs gas_v4i;
typedef __attribute__((address_space(3))) v4i_vs las_v4i;
__device__ __forceinline__ void async_wait0() {
#  if __has_builtin(__builtin_amdgcn_s_wait_asynccnt)
    __builtin_amdgcn_s_wait_asynccnt(0);
#  else
    asm volatile("s_wait_asynccnt 0x0" ::: "memory");
#  endif
}
#endif

// Copy n f32 (n%4==0, 16B aligned) from global to an LDS buffer.
__device__ __forceinline__ void stage_row_f32(
        const float* __restrict__ src, float* __restrict__ lds_dst,
        int n, int tid, int nthr) {
#ifdef HAVE_ASYNC_LDS
    for (int c = tid * 4; c < n; c += nthr * 4)
        __builtin_amdgcn_global_load_async_to_lds_b128(
            (gas_v4i*)(src + c), (las_v4i*)(lds_dst + c), 0, 0);
    async_wait0();
#else
    for (int c = tid; c < n; c += nthr) lds_dst[c] = src[c];
#endif
}

// ---------------------------------------------------------------------------
// Elementwise cast
// ---------------------------------------------------------------------------
__global__ __launch_bounds__(256) void cast_f32_to_f16_kernel(
        const float* __restrict__ in, _Float16* __restrict__ out, int n) {
    int i = blockIdx.x * 256 + threadIdx.x;
    if (i < n) out[i] = (_Float16)in[i];
}

// w[K][N] (row-major) -> wt[N][K] f16, tiled through LDS so both global
// accesses are coalesced. Grid: (K/32)*(N/32) blocks of 256 (32x8) threads.
__global__ __launch_bounds__(256) void transpose_cast_kernel(
        const float* __restrict__ w, _Float16* __restrict__ wt, int K, int N) {
    __shared__ float tile[32][33];
    int kt = K >> 5;
    int bk = blockIdx.x % kt;
    int bn = blockIdx.x / kt;
    int tx = threadIdx.x & 31, ty = threadIdx.x >> 5;
#pragma unroll
    for (int i = 0; i < 4; ++i)
        tile[ty + i * 8][tx] = w[(size_t)(bk * 32 + ty + i * 8) * N + bn * 32 + tx];
    __syncthreads();
#pragma unroll
    for (int i = 0; i < 4; ++i)
        wt[(size_t)(bn * 32 + ty + i * 8) * K + bk * 32 + tx] =
            (_Float16)tile[tx][ty + i * 8];
}

// ---------------------------------------------------------------------------
// WMMA GEMM:  out[M][N] = X[M][K] (f16) * Wt[N][K]^T (f16) + bias, f32 accum.
// One wave -> 16x64 tile. Ping-pong register double buffering with a 2x
// unrolled K-loop: no inter-buffer copies, loads overlap WMMAs.
// ---------------------------------------------------------------------------
__global__ __launch_bounds__(256) void gemm_f16_kernel(
        const _Float16* __restrict__ X, const _Float16* __restrict__ Wt,
        const float* __restrict__ bias, float* __restrict__ out,
        int M, int N, int K) {
    int wave   = blockIdx.x * (blockDim.x >> 5) + (threadIdx.x >> 5);
    int ntiles = N >> 6;
    int mt = wave / ntiles;
    int nt = wave % ntiles;
    if (mt * 16 >= M) return;                 // whole wave exits uniformly
    int lane = threadIdx.x & 31;
    int lh = lane >> 4, ln = lane & 15;
    int m0 = mt * 16, n0 = nt * 64;

    const _Float16* ap = X + (size_t)(m0 + ln) * K;
    const _Float16* bp = Wt + (size_t)(n0 + ln) * K;
    auto ldA = [&](int kk) {
        return cat8(*(const v8h*)(ap + kk + lh * 8),
                    *(const v8h*)(ap + kk + 16 + lh * 8));
    };
    auto ldB = [&](int kk, int t) {
        const _Float16* p = bp + (size_t)t * 16 * K + kk + lh * 8;
        return cat8(*(const v8h*)p, *(const v8h*)(p + 16));
    };

    v8f acc[4] = {};
    v16h a0, a1, b0[4], b1[4];
    a0 = ldA(0);
#pragma unroll
    for (int t = 0; t < 4; ++t) b0[t] = ldB(0, t);

    for (int kk = 0; kk < K; kk += 64) {
        bool has1 = (kk + 32) < K;
        if (has1) {
            a1 = ldA(kk + 32);
#pragma unroll
            for (int t = 0; t < 4; ++t) b1[t] = ldB(kk + 32, t);
        }
#pragma unroll
        for (int t = 0; t < 4; ++t)
            acc[t] = __builtin_amdgcn_wmma_f32_16x16x32_f16(
                         false, a0, false, b0[t], (short)0, acc[t], false, false);
        if (kk + 64 < K) {
            a0 = ldA(kk + 64);
#pragma unroll
            for (int t = 0; t < 4; ++t) b0[t] = ldB(kk + 64, t);
        }
        if (has1) {
#pragma unroll
            for (int t = 0; t < 4; ++t)
                acc[t] = __builtin_amdgcn_wmma_f32_16x16x32_f16(
                             false, a1, false, b1[t], (short)0, acc[t], false, false);
        }
    }

#pragma unroll
    for (int t = 0; t < 4; ++t) {
        int col = n0 + t * 16 + ln;
        float bv = bias[col];
#pragma unroll
        for (int r = 0; r < 8; ++r) {
            int row = m0 + lh * 8 + r;        // C layout: VGPR r -> M = r + lh*8
            out[(size_t)row * N + col] = acc[t][r] + bv;
        }
    }
}

// ---------------------------------------------------------------------------
// Q build: rmsnorm(q_raw, gq) + RoPE (frames 6..13) -> Qh[head][tok][128] f16
// ---------------------------------------------------------------------------
__global__ __launch_bounds__(256) void build_q_kernel(
        const float* __restrict__ q_raw, const float* __restrict__ gq,
        const float* __restrict__ freqs, _Float16* __restrict__ Qh) {
    int tok = blockIdx.x;
    __shared__ __align__(16) float row[DIM];
    __shared__ float red[256];
    stage_row_f32(q_raw + (size_t)tok * DIM, row, DIM, threadIdx.x, 256);
    __syncthreads();
    float ss = 0.f;
    for (int c = threadIdx.x; c < DIM; c += 256) { float v = row[c]; ss += v * v; }
    red[threadIdx.x] = ss; __syncthreads();
    for (int s = 128; s > 0; s >>= 1) {
        if (threadIdx.x < s) red[threadIdx.x] += red[threadIdx.x + s];
        __syncthreads();
    }
    float rms = rsqrtf(red[0] / (float)DIM + 1e-6f);
    int frame = 6 + tok / FS;
    int idx = tok % FS;
    int hh = idx / 20, ww = idx % 20;
    for (int p = threadIdx.x; p < DIM / 2; p += 256) {
        int head = p >> 6, ci = p & 63;
        float ang = (ci < C0)      ? freqs[frame * C_HALF + ci]
                  : (ci < C0 + C1) ? freqs[hh * C_HALF + ci]
                                   : freqs[ww * C_HALF + ci];
        float cs = __cosf(ang), sn = __sinf(ang);
        float a = row[2 * p]     * rms * gq[2 * p];
        float b = row[2 * p + 1] * rms * gq[2 * p + 1];
        size_t o = ((size_t)head * SEQ + tok) * D_H + 2 * ci;
        Qh[o]     = (_Float16)(a * cs - b * sn);
        Qh[o + 1] = (_Float16)(a * sn + b * cs);
    }
}

// ---------------------------------------------------------------------------
// K build: concat(cache[0:2f], 5:1 compress x2, cache[12f:14f], rmsnorm(k_new))
// + RoPE frames 0..13; pad rows [4200,4224) with zeros. -> Kh[head][key][128]
// ---------------------------------------------------------------------------
__global__ __launch_bounds__(256) void build_k_kernel(
        const float* __restrict__ cache_k, const float* __restrict__ k_raw,
        const float* __restrict__ gk, const float* __restrict__ w_memk,
        const float* __restrict__ freqs, _Float16* __restrict__ Kh) {
    int j = blockIdx.x;                       // 0..KEYP-1 (uniform per block)
    if (j >= KEYS) {                          // zero pad
        for (int c = threadIdx.x; c < DIM; c += 256) {
            int head = c >> 7, d = c & 127;
            Kh[((size_t)head * KEYP + j) * D_H + d] = (_Float16)0.f;
        }
        return;
    }
    __shared__ __align__(16) float row[DIM];
    __shared__ float red[256];
    bool need_norm = (j >= 1800);
    if (j < 600) {
        stage_row_f32(cache_k + (size_t)j * DIM, row, DIM, threadIdx.x, 256);
    } else if (j < 1200) {
        int base = (j < 900) ? (600 + 5 * (j - 600)) : (2100 + 5 * (j - 900));
        for (int c = threadIdx.x; c < DIM; c += 256) {
            float acc = 0.f;
#pragma unroll
            for (int jj = 0; jj < 5; ++jj)
                acc += cache_k[(size_t)(base + jj) * DIM + c] * w_memk[c * 5 + jj];
            row[c] = acc;
        }
    } else if (j < 1800) {
        stage_row_f32(cache_k + (size_t)(3600 + (j - 1200)) * DIM, row, DIM, threadIdx.x, 256);
    } else {
        stage_row_f32(k_raw + (size_t)(j - 1800) * DIM, row, DIM, threadIdx.x, 256);
    }
    __syncthreads();
    float rms = 1.f;
    if (need_norm) {
        float ss = 0.f;
        for (int c = threadIdx.x; c < DIM; c += 256) ss += row[c] * row[c];
        red[threadIdx.x] = ss; __syncthreads();
        for (int s = 128; s > 0; s >>= 1) {
            if (threadIdx.x < s) red[threadIdx.x] += red[threadIdx.x + s];
            __syncthreads();
        }
        rms = rsqrtf(red[0] / (float)DIM + 1e-6f);
    }
    int frame = j / FS;
    int idx = j % FS;
    int hh = idx / 20, ww = idx % 20;
    for (int p = threadIdx.x; p < DIM / 2; p += 256) {
        int head = p >> 6, ci = p & 63;
        float ang = (ci < C0)      ? freqs[frame * C_HALF + ci]
                  : (ci < C0 + C1) ? freqs[hh * C_HALF + ci]
                                   : freqs[ww * C_HALF + ci];
        float cs = __cosf(ang), sn = __sinf(ang);
        float sa = need_norm ? rms * gk[2 * p]     : 1.f;
        float sb = need_norm ? rms * gk[2 * p + 1] : 1.f;
        float a = row[2 * p] * sa;
        float b = row[2 * p + 1] * sb;
        size_t o = ((size_t)head * KEYP + j) * D_H + 2 * ci;
        Kh[o]     = (_Float16)(a * cs - b * sn);
        Kh[o + 1] = (_Float16)(a * sn + b * cs);
    }
}

// ---------------------------------------------------------------------------
// V build (same concat, no norm/rope), stored TRANSPOSED: Vt[head][d][key]
// so the PV matmul B-fragments are contiguous b128 loads.
// ---------------------------------------------------------------------------
__global__ __launch_bounds__(256) void build_v_kernel(
        const float* __restrict__ cache_v, const float* __restrict__ v_raw,
        const float* __restrict__ w_memk, _Float16* __restrict__ Vt) {
    int j = blockIdx.x;
    for (int c = threadIdx.x; c < DIM; c += 256) {
        float val;
        if (j >= KEYS) {
            val = 0.f;
        } else if (j < 600) {
            val = cache_v[(size_t)j * DIM + c];
        } else if (j < 1200) {
            int base = (j < 900) ? (600 + 5 * (j - 600)) : (2100 + 5 * (j - 900));
            float acc = 0.f;
#pragma unroll
            for (int jj = 0; jj < 5; ++jj)
                acc += cache_v[(size_t)(base + jj) * DIM + c] * w_memk[c * 5 + jj];
            val = acc;
        } else if (j < 1800) {
            val = cache_v[(size_t)(3600 + (j - 1200)) * DIM + c];
        } else {
            val = v_raw[(size_t)(j - 1800) * DIM + c];
        }
        int head = c >> 7, d = c & 127;
        Vt[((size_t)head * D_H + d) * KEYP + j] = (_Float16)val;
    }
}

// ---------------------------------------------------------------------------
// Flash attention: one wave per (head, 16-query tile). Online softmax,
// QK^T and PV via v_wmma_f32_16x16x32_f16, P staged through per-wave LDS.
// K fragments are batch-loaded (one clause) before the QK WMMA chain; V
// fragments are issued before the softmax VALU work so loads overlap it.
// ---------------------------------------------------------------------------
__global__ __launch_bounds__(256) void attn_kernel(
        const _Float16* __restrict__ Qh, const _Float16* __restrict__ Kh,
        const _Float16* __restrict__ Vt, float* __restrict__ O) {
    __shared__ _Float16 lds_p[8][16 * 32];    // per-wave 16x32 P tile
    int wv   = threadIdx.x >> 5;
    int task = blockIdx.x * 8 + wv;           // 12 heads * 150 q-tiles = 1800
    int h  = task / (SEQ / 16);
    int qt = task % (SEQ / 16);
    int lane = threadIdx.x & 31;
    int lh = lane >> 4, ln = lane & 15;

    // Q fragments (A-layout), 16 rows x 128 d, 4 chunks of K=32
    v16h qf[4];
    const _Float16* qp = Qh + ((size_t)h * SEQ + qt * 16 + ln) * D_H;
#pragma unroll
    for (int c = 0; c < 4; ++c)
        qf[c] = cat8(*(const v8h*)(qp + c * 32 + lh * 8),
                     *(const v8h*)(qp + c * 32 + 16 + lh * 8));

    v8f oacc[8] = {};
    float mrow[8], lsum[8];
#pragma unroll
    for (int r = 0; r < 8; ++r) { mrow[r] = -3.0e38f; lsum[r] = 0.f; }

    const _Float16* kbase = Kh + (size_t)h * KEYP * D_H;
    const _Float16* vbase = Vt + (size_t)h * D_H * KEYP;

    for (int kb = 0; kb < KEYP; kb += 32) {
        // ---- batch-load all 8 K fragments (two 16-key tiles x 4 d-chunks) ----
        v16h kf0[4], kf1[4];
#pragma unroll
        for (int c = 0; c < 4; ++c) {
            const _Float16* kp0 = kbase + (size_t)(kb + ln) * D_H + c * 32 + lh * 8;
            const _Float16* kp1 = kp0 + (size_t)16 * D_H;
            kf0[c] = cat8(*(const v8h*)kp0, *(const v8h*)(kp0 + 16));
            kf1[c] = cat8(*(const v8h*)kp1, *(const v8h*)(kp1 + 16));
        }
        // ---- S = Q K^T ----
        v8f s0 = {}, s1 = {};
#pragma unroll
        for (int c = 0; c < 4; ++c) {
            s0 = __builtin_amdgcn_wmma_f32_16x16x32_f16(false, qf[c], false, kf0[c], (short)0, s0, false, false);
            s1 = __builtin_amdgcn_wmma_f32_16x16x32_f16(false, qf[c], false, kf1[c], (short)0, s1, false, false);
        }

        // ---- issue V fragment loads early: independent of softmax math ----
        v16h vf[8];
#pragma unroll
        for (int t = 0; t < 8; ++t) {
            const _Float16* vp = vbase + (size_t)(t * 16 + ln) * KEYP + kb + lh * 8;
            vf[t] = cat8(*(const v8h*)vp, *(const v8h*)(vp + 16));
        }
        if (kb + 32 < KEYP)                   // global_prefetch_b8 of next K block
            __builtin_prefetch(kbase + (size_t)(kb + 32 + ln) * D_H, 0, 1);

        // ---- online softmax (rows live across the 16 lanes of each half) ----
        int col0 = kb + ln, col1 = kb + 16 + ln;
        float p0[8], p1[8], mx[8];
#pragma unroll
        for (int r = 0; r < 8; ++r) {
            float v0 = s0[r] * ATT_SCALE; if (col0 >= KEYS) v0 = -1.0e9f;
            float v1 = s1[r] * ATT_SCALE; if (col1 >= KEYS) v1 = -1.0e9f;
            p0[r] = v0; p1[r] = v1;
            mx[r] = fmaxf(v0, v1);
            mx[r] = fmaxf(mx[r], __shfl_xor(mx[r], 1, 32));
            mx[r] = fmaxf(mx[r], __shfl_xor(mx[r], 2, 32));
            mx[r] = fmaxf(mx[r], __shfl_xor(mx[r], 4, 32));
            mx[r] = fmaxf(mx[r], __shfl_xor(mx[r], 8, 32));
        }
#pragma unroll
        for (int r = 0; r < 8; ++r) {
            float mnew  = fmaxf(mrow[r], mx[r]);
            float alpha = __expf(mrow[r] - mnew);
            mrow[r] = mnew;
            float e0 = __expf(p0[r] - mnew);
            float e1 = __expf(p1[r] - mnew);
            p0[r] = e0; p1[r] = e1;
            float rs = e0 + e1;
            rs += __shfl_xor(rs, 1, 32);
            rs += __shfl_xor(rs, 2, 32);
            rs += __shfl_xor(rs, 4, 32);
            rs += __shfl_xor(rs, 8, 32);
            lsum[r] = lsum[r] * alpha + rs;
#pragma unroll
            for (int t = 0; t < 8; ++t) oacc[t][r] *= alpha;
        }

        // ---- C-layout P -> A-layout via per-wave LDS (LDS is in-order/wave) ----
        _Float16* lp = lds_p[wv];
#pragma unroll
        for (int r = 0; r < 8; ++r) {
            lp[(lh * 8 + r) * 32 + ln]      = (_Float16)p0[r];
            lp[(lh * 8 + r) * 32 + 16 + ln] = (_Float16)p1[r];
        }
        asm volatile("s_wait_dscnt 0x0" ::: "memory");
        v16h pf = cat8(*(const v8h*)(lp + ln * 32 + lh * 8),
                       *(const v8h*)(lp + ln * 32 + 16 + lh * 8));

        // ---- O += P V (contraction over 32 keys, 8 d-tiles) ----
#pragma unroll
        for (int t = 0; t < 8; ++t)
            oacc[t] = __builtin_amdgcn_wmma_f32_16x16x32_f16(false, pf, false, vf[t], (short)0, oacc[t], false, false);
    }

    // ---- epilogue: normalize rows, write O[token][head*128+d] ----
#pragma unroll
    for (int r = 0; r < 8; ++r) {
        float inv = 1.f / lsum[r];
        int row = qt * 16 + lh * 8 + r;
#pragma unroll
        for (int t = 0; t < 8; ++t)
            O[(size_t)row * DIM + h * D_H + t * 16 + ln] = oacc[t][r] * inv;
    }
}

// ---------------------------------------------------------------------------
// Host orchestration
// ---------------------------------------------------------------------------
extern "C" void kernel_launch(void* const* d_in, const int* in_sizes, int n_in,
                              void* d_out, int out_size, void* d_ws, size_t ws_size,
                              hipStream_t stream) {
    const float* x       = (const float*)d_in[0];
    const float* cache_k = (const float*)d_in[1];
    const float* cache_v = (const float*)d_in[2];
    const float* freqs   = (const float*)d_in[3];
    const float* wq = (const float*)d_in[6];   const float* bq = (const float*)d_in[7];
    const float* wk = (const float*)d_in[8];   const float* bk = (const float*)d_in[9];
    const float* wv = (const float*)d_in[10];  const float* bv = (const float*)d_in[11];
    const float* wo = (const float*)d_in[12];  const float* bo = (const float*)d_in[13];
    const float* gq = (const float*)d_in[14];  const float* gk = (const float*)d_in[15];
    const float* w_memk = (const float*)d_in[16];

    size_t off = 0;
    auto alloc = [&](size_t bytes) {
        void* p = (char*)d_ws + off;
        off += (bytes + 255) & ~(size_t)255;
        return p;
    };
    _Float16* xh   = (_Float16*)alloc((size_t)SEQ * DIM * 2);
    _Float16* wqt  = (_Float16*)alloc((size_t)DIM * DIM * 2);
    _Float16* wkt  = (_Float16*)alloc((size_t)DIM * DIM * 2);
    _Float16* wvt  = (_Float16*)alloc((size_t)DIM * DIM * 2);
    _Float16* wot  = (_Float16*)alloc((size_t)DIM * DIM * 2);
    float*    q_raw = (float*)alloc((size_t)SEQ * DIM * 4);
    float*    k_raw = (float*)alloc((size_t)SEQ * DIM * 4);
    float*    v_raw = (float*)alloc((size_t)SEQ * DIM * 4);
    _Float16* Qh   = (_Float16*)alloc((size_t)N_H * SEQ * D_H * 2);
    _Float16* Kh   = (_Float16*)alloc((size_t)N_H * KEYP * D_H * 2);
    _Float16* Vt   = (_Float16*)alloc((size_t)N_H * D_H * KEYP * 2);
    float*    Obuf = (float*)alloc((size_t)SEQ * DIM * 4);
    _Float16* Oh   = (_Float16*)alloc((size_t)SEQ * DIM * 2);

    const int nx = SEQ * DIM;            // 3,686,400
    cast_f32_to_f16_kernel<<<(nx + 255) / 256, 256, 0, stream>>>(x, xh, nx);

    const int tr_blocks = (DIM / 32) * (DIM / 32);     // 2304
    transpose_cast_kernel<<<tr_blocks, 256, 0, stream>>>(wq, wqt, DIM, DIM);
    transpose_cast_kernel<<<tr_blocks, 256, 0, stream>>>(wk, wkt, DIM, DIM);
    transpose_cast_kernel<<<tr_blocks, 256, 0, stream>>>(wv, wvt, DIM, DIM);
    transpose_cast_kernel<<<tr_blocks, 256, 0, stream>>>(wo, wot, DIM, DIM);

    const int gemm_tasks  = (SEQ / 16) * (DIM / 64);   // 3600 waves
    const int gemm_blocks = (gemm_tasks + 7) / 8;      // 450
    gemm_f16_kernel<<<gemm_blocks, 256, 0, stream>>>(xh, wqt, bq, q_raw, SEQ, DIM, DIM);
    gemm_f16_kernel<<<gemm_blocks, 256, 0, stream>>>(xh, wkt, bk, k_raw, SEQ, DIM, DIM);
    gemm_f16_kernel<<<gemm_blocks, 256, 0, stream>>>(xh, wvt, bv, v_raw, SEQ, DIM, DIM);

    build_q_kernel<<<SEQ, 256, 0, stream>>>(q_raw, gq, freqs, Qh);
    build_k_kernel<<<KEYP, 256, 0, stream>>>(cache_k, k_raw, gk, w_memk, freqs, Kh);
    build_v_kernel<<<KEYP, 256, 0, stream>>>(cache_v, v_raw, w_memk, Vt);

    const int attn_blocks = (N_H * (SEQ / 16)) / 8;    // 225
    attn_kernel<<<attn_blocks, 256, 0, stream>>>(Qh, Kh, Vt, Obuf);

    cast_f32_to_f16_kernel<<<(nx + 255) / 256, 256, 0, stream>>>(Obuf, Oh, nx);
    gemm_f16_kernel<<<gemm_blocks, 256, 0, stream>>>(Oh, wot, bo, (float*)d_out, SEQ, DIM, DIM);
}
